// FeatureRefinement_50225347559626
// MI455X (gfx1250) — matrix operand order repaired and checked by reading, not compiled, not run on hardware
//
#include <hip/hip_runtime.h>
#include <math.h>

typedef __attribute__((ext_vector_type(16))) _Float16 v16h;
typedef __attribute__((ext_vector_type(4)))  _Float16 v4h;
typedef __attribute__((ext_vector_type(8)))  float    v8f;

#define BS 16
#define VL 1024
#define QL 64
#define DM 1024
#define NN 1024          // output cols
#define KK 1024          // reduction depth of big GEMM
#define MM (BS * VL)     // 16384 rows
#define NEG_INF (-1e30f)

// ---------------------------------------------------------------------------
// Kernel 1: per-batch sentence feature (weighted pool with softmax) + its norm
// ---------------------------------------------------------------------------
__global__ __launch_bounds__(256)
void k_sentence(const float* __restrict__ qf, const float* __restrict__ qmask,
                const float* __restrict__ pool_w,
                float* __restrict__ sentence, float* __restrict__ sn) {
  const int b = blockIdx.x;
  const int tid = threadIdx.x, lane = tid & 31, wave = tid >> 5;
  __shared__ float salpha[QL];
  __shared__ float red[8];

  for (int q = wave; q < QL; q += 8) {
    const float* row = qf + ((size_t)b * QL + q) * DM;
    float p = 0.f;
    for (int k = lane; k < DM; k += 32) p += row[k] * pool_w[k];
    for (int off = 16; off > 0; off >>= 1) p += __shfl_xor(p, off, 32);
    if (lane == 0) salpha[q] = p + (1.f - qmask[b * QL + q]) * NEG_INF;
  }
  __syncthreads();

  float mx = -INFINITY;
  for (int q = 0; q < QL; q++) mx = fmaxf(mx, salpha[q]);
  float sum = 0.f;
  for (int q = 0; q < QL; q++) sum += expf(salpha[q] - mx);
  float inv = 1.f / sum;
  float myv = (tid < QL) ? expf(salpha[tid] - mx) * inv : 0.f;
  __syncthreads();
  if (tid < QL) salpha[tid] = myv;
  __syncthreads();

  float s2 = 0.f;
  #pragma unroll
  for (int i = 0; i < DM / 256; i++) {
    const int d = tid + i * 256;
    float acc = 0.f;
    for (int q = 0; q < QL; q++)
      acc += salpha[q] * qf[((size_t)b * QL + q) * DM + d];
    sentence[(size_t)b * DM + d] = acc;
    s2 += acc * acc;
  }
  for (int off = 16; off > 0; off >>= 1) s2 += __shfl_xor(s2, off, 32);
  if (lane == 0) red[wave] = s2;
  __syncthreads();
  if (tid == 0) {
    float t = 0.f;
    for (int w = 0; w < 8; w++) t += red[w];
    sn[b] = fmaxf(sqrtf(t), 1e-8f);
  }
}

// ---------------------------------------------------------------------------
// Kernel 2: r1[j] = sim_w^T @ W1[:,j];  c3[j] = cor_q * (cor_v_w^T @ W3[:,j])
// ---------------------------------------------------------------------------
__global__ __launch_bounds__(256)
void k_vec(const float* __restrict__ sim_w, const float* __restrict__ cor_v_w,
           const float* __restrict__ cor_q_w, const float* __restrict__ mixer_w,
           float* __restrict__ r1, float* __restrict__ c3) {
  const int j = blockIdx.x * 256 + threadIdx.x;
  const float* W1 = mixer_w + (size_t)1 * DM * NN;
  const float* W3 = mixer_w + (size_t)3 * DM * NN;
  float a = 0.f, c = 0.f;
  for (int d = 0; d < DM; d++) {
    a += sim_w[d]   * W1[(size_t)d * NN + j];
    c += cor_v_w[d] * W3[(size_t)d * NN + j];
  }
  r1[j] = a;
  c3[j] = c * cor_q_w[0];
}

// ---------------------------------------------------------------------------
// Kernel 3: base[b,j] = sentence[b,:] @ W2[:,j] + c3[j] + bias[j]
// ---------------------------------------------------------------------------
__global__ __launch_bounds__(256)
void k_base(const float* __restrict__ sentence, const float* __restrict__ mixer_w,
            const float* __restrict__ mixer_b, const float* __restrict__ c3,
            float* __restrict__ base) {
  const int j = blockIdx.x * 256 + threadIdx.x;
  const int b = blockIdx.y;
  const float* W2 = mixer_w + (size_t)2 * DM * NN;
  const float* s  = sentence + (size_t)b * DM;
  float acc = 0.f;
  for (int d = 0; d < DM; d++) acc += s[d] * W2[(size_t)d * NN + j];
  base[(size_t)b * NN + j] = acc + c3[j] + mixer_b[j];
}

// ---------------------------------------------------------------------------
// Kernel 4: pre-convert + pre-swizzle W0 (f32 KxN row-major) into f16 WMMA
// B-fragments: wsB[ntile(64)][kchunk(32)][lane(32)][half(16)]
//   lane = (n&15) | (((klocal>>3)&1)<<4) ; half = (klocal&7) | ((klocal>>4)<<3)
// ---------------------------------------------------------------------------
__global__ __launch_bounds__(256)
void k_packB(const float* __restrict__ W0, _Float16* __restrict__ wsB) {
  const int f = (blockIdx.x * 256 + threadIdx.x) * 4;  // flat half index
  const int h0    = f & 15;           // 0,4,8,12
  const int lane  = (f >> 4) & 31;
  const int kc    = (f >> 9) & 31;
  const int ntile = f >> 14;
  const int n   = ntile * 16 + (lane & 15);
  const int kHi = (lane >> 4) << 3;
  v4h t;
  #pragma unroll
  for (int j = 0; j < 4; j++) {
    const int k = kc * 32 + (((h0 + j) & 8) << 1) + kHi + ((h0 + j) & 7);
    t[j] = (_Float16)W0[(size_t)k * NN + n];
  }
  *(v4h*)(wsB + f) = t;
}

// ---------------------------------------------------------------------------
// Kernel 5: fused WMMA GEMM + cosine-sim + epilogue (LDS-free GEMM path):
//   out = relu(video @ W0 + sim*r1 + base),
//   sim computed in-flight from the f32 A values already in registers.
//   grid (NN/64, MM/128); 8 waves/block; wave = 16(M) x 64(N); BK = 32.
// ---------------------------------------------------------------------------
__global__ __launch_bounds__(256)
void k_gemm(const float* __restrict__ video, const _Float16* __restrict__ wsB,
            const float* __restrict__ vmask, const float* __restrict__ sentence,
            const float* __restrict__ sn, const float* __restrict__ r1,
            const float* __restrict__ base, float* __restrict__ out) {
  const int tid  = threadIdx.x;
  const int lane = tid & 31;
  const int wave = tid >> 5;
  const int rowBase = blockIdx.y * 128;
  const int colBase = blockIdx.x * 64;
  const int b = rowBase >> 10;          // 128 | VL: one batch per block

  // Stage this batch's sentence vector (4 KB) into LDS once.
  __shared__ float sSent[DM];
  ((float4*)sSent)[tid] = ((const float4*)(sentence + (size_t)b * DM))[tid];
  __syncthreads();

  // A-fragment addressing per ISA layout: lane l holds row (l&15);
  // K-halves 0-7 = kb+d..kb+d+7, halves 8-15 = kb+16+d..+7, d = 8*(l>=16)
  const int rowA = rowBase + wave * 16 + (lane & 15);
  const int d8   = (lane >> 4) << 3;
  const float* aRow  = video + (size_t)rowA * KK + d8;
  const float* sLane = sSent + d8;

  // B-fragment base: frag(ntileGlobal, kc) at (ntileGlobal*32 + kc)*512 + lane*16
  const _Float16* bBase = wsB + (size_t)(blockIdx.x * 4) * 32 * 512 + lane * 16;

  v8f acc0 = {}, acc1 = {}, acc2 = {}, acc3 = {};
  float dotp = 0.f, nrmp = 0.f;

  for (int kc = 0; kc < KK / 32; kc++) {
    const float* ap = aRow + kc * 32;
    const float4 f0 = *(const float4*)(ap);
    const float4 f1 = *(const float4*)(ap + 4);
    const float4 f2 = *(const float4*)(ap + 16);
    const float4 f3 = *(const float4*)(ap + 20);

    // cosine-sim partials on exact f32 data (LDS-resident sentence)
    const float* sp = sLane + kc * 32;
    const float4 s0 = *(const float4*)(sp);
    const float4 s1 = *(const float4*)(sp + 4);
    const float4 s2 = *(const float4*)(sp + 16);
    const float4 s3 = *(const float4*)(sp + 20);
    dotp += f0.x*s0.x + f0.y*s0.y + f0.z*s0.z + f0.w*s0.w
          + f1.x*s1.x + f1.y*s1.y + f1.z*s1.z + f1.w*s1.w
          + f2.x*s2.x + f2.y*s2.y + f2.z*s2.z + f2.w*s2.w
          + f3.x*s3.x + f3.y*s3.y + f3.z*s3.z + f3.w*s3.w;
    nrmp += f0.x*f0.x + f0.y*f0.y + f0.z*f0.z + f0.w*f0.w
          + f1.x*f1.x + f1.y*f1.y + f1.z*f1.z + f1.w*f1.w
          + f2.x*f2.x + f2.y*f2.y + f2.z*f2.z + f2.w*f2.w
          + f3.x*f3.x + f3.y*f3.y + f3.z*f3.z + f3.w*f3.w;

    v16h av;
    av[0]  = (_Float16)f0.x; av[1]  = (_Float16)f0.y;
    av[2]  = (_Float16)f0.z; av[3]  = (_Float16)f0.w;
    av[4]  = (_Float16)f1.x; av[5]  = (_Float16)f1.y;
    av[6]  = (_Float16)f1.z; av[7]  = (_Float16)f1.w;
    av[8]  = (_Float16)f2.x; av[9]  = (_Float16)f2.y;
    av[10] = (_Float16)f2.z; av[11] = (_Float16)f2.w;
    av[12] = (_Float16)f3.x; av[13] = (_Float16)f3.y;
    av[14] = (_Float16)f3.z; av[15] = (_Float16)f3.w;

    const _Float16* bp = bBase + (size_t)kc * 512;
    const v16h b0 = *(const v16h*)(bp);
    const v16h b1 = *(const v16h*)(bp + 32 * 512);
    const v16h b2 = *(const v16h*)(bp + 2 * 32 * 512);
    const v16h b3 = *(const v16h*)(bp + 3 * 32 * 512);

    acc0 = __builtin_amdgcn_wmma_f32_16x16x32_f16(false, av, false, b0, (short)0, acc0, false, false);
    acc1 = __builtin_amdgcn_wmma_f32_16x16x32_f16(false, av, false, b1, (short)0, acc1, false, false);
    acc2 = __builtin_amdgcn_wmma_f32_16x16x32_f16(false, av, false, b2, (short)0, acc2, false, false);
    acc3 = __builtin_amdgcn_wmma_f32_16x16x32_f16(false, av, false, b3, (short)0, acc3, false, false);
  }

  // Fold the two half-row partials (lane l covers K {0..7,16..23} mod 32,
  // lane l+16 the complement) -> full dot/norm for row (lane&15).
  dotp += __shfl_xor(dotp, 16, 32);
  nrmp += __shfl_xor(nrmp, 16, 32);
  const float vn   = fmaxf(sqrtf(nrmp), 1e-8f);
  const float simv = dotp / (vn * sn[b]) + logf(vmask[rowA] + 1e-45f);

  // Fused epilogue. C/D layout: VGPR r, lane l -> M = r + 8*(l>=16), N = l%16
  const int mOff = rowBase + wave * 16 + d8;
  const int nSub = lane & 15;
  v8f accs[4] = {acc0, acc1, acc2, acc3};
  #pragma unroll
  for (int nt = 0; nt < 4; nt++) {
    const int col = colBase + nt * 16 + nSub;
    const float r1c = r1[col];
    #pragma unroll
    for (int r = 0; r < 8; r++) {
      const int rl   = ((lane >> 4) << 3) + r;       // local row 0..15
      const float sr = __shfl(simv, rl, 32);         // sim of that row
      const int row  = mOff + r;
      const float val = accs[nt][r] + sr * r1c + base[(size_t)b * NN + col];
      out[(size_t)row * NN + col] = fmaxf(val, 0.f);
    }
  }
}

// ---------------------------------------------------------------------------
extern "C" void kernel_launch(void* const* d_in, const int* in_sizes, int n_in,
                              void* d_out, int out_size, void* d_ws, size_t ws_size,
                              hipStream_t stream) {
  const float* video    = (const float*)d_in[0];
  const float* qfeat    = (const float*)d_in[1];
  const float* vmask    = (const float*)d_in[2];
  const float* qmask    = (const float*)d_in[3];
  const float* sim_w    = (const float*)d_in[4];
  const float* cor_v_w  = (const float*)d_in[5];
  const float* cor_q_w  = (const float*)d_in[6];
  const float* pool_w   = (const float*)d_in[7];
  const float* mixer_w  = (const float*)d_in[8];
  const float* mixer_b  = (const float*)d_in[9];
  float* out = (float*)d_out;

  float* ws       = (float*)d_ws;
  float* sentence = ws;                       // 16*1024
  float* sn       = ws + 16384;               // 16
  float* r1       = ws + 16400;               // 1024
  float* c3       = ws + 17424;               // 1024
  float* base     = ws + 18448;               // 16*1024 (ends 34832)
  _Float16* wsB   = (_Float16*)(ws + 34848);  // 1M halves = 2 MB, 32B aligned

  k_sentence<<<BS, 256, 0, stream>>>(qfeat, qmask, pool_w, sentence, sn);
  k_vec<<<NN / 256, 256, 0, stream>>>(sim_w, cor_v_w, cor_q_w, mixer_w, r1, c3);
  k_base<<<dim3(NN / 256, BS), 256, 0, stream>>>(sentence, mixer_w, mixer_b, c3, base);
  k_packB<<<(64 * 32 * 32 * 16) / (4 * 256), 256, 0, stream>>>(mixer_w, wsB);
  k_gemm<<<dim3(NN / 64, MM / 128), 256, 0, stream>>>(video, wsB, vmask, sentence,
                                                      sn, r1, base, out);
}
